// GEM_model_22548578304801
// MI455X (gfx1250) — matrix operand order repaired and checked by reading, not compiled, not run on hardware
//
#include <hip/hip_runtime.h>
#include <hip/hip_bf16.h>
#include <math.h>

typedef __attribute__((ext_vector_type(16))) __bf16 v16bf;
typedef __attribute__((ext_vector_type(8)))  float  v8f;

#define D_IN   50176
#define H1     512
#define H2     256
#define NOUT   16
#define BATCH  512
#define SPLITS 16
#define KCHUNK (D_IN / SPLITS)   /* 3136, divisible by 32 */

/* workspace layout (bytes)
   [0, 16MB)        f32 partials[SPLITS][512][512]
   [16MB, +512KB)   h1 bf16 [512][512]
   [.., +512KB)     h2 f32  [512][256]                               */
#define WS_PART_OFF 0
#define WS_H1_OFF   ((size_t)SPLITS * BATCH * H1 * sizeof(float))
#define WS_H2_OFF   (WS_H1_OFF + (size_t)BATCH * H1 * sizeof(__bf16))

/* native bf16 conversion: lets the backend pick v_cvt (packed) forms */
__device__ __forceinline__ __bf16 f2bf(float f) { return (__bf16)f; }

/* A fragment (16x32 bf16, M x K): lanes 0-15 row M=lane carry K = {hi*16..hi*16+15},
   matching the contiguous B order. LDS tile is [128][32] row-major. */
__device__ __forceinline__ v16bf load_frag_A(const __bf16* sA, int m, int hi) {
  union { uint4 u[2]; v16bf v; } f;
  f.u[0] = *(const uint4*)(sA + m * 32 + hi * 16);
  f.u[1] = *(const uint4*)(sA + m * 32 + hi * 16 + 8);
  return f.v;
}
/* B fragment (32x16 bf16, K x N): lane = (hi*16 + n%16); element j is K=kg+j.
   LDS tile stored K-transposed as [n][k] so this is two contiguous b128s. */
__device__ __forceinline__ v16bf load_frag_B(const __bf16* sBt, int n, int kg) {
  union { uint4 u[2]; v16bf v; } f;
  const uint4* p = (const uint4*)(sBt + n * 32 + kg);
  f.u[0] = p[0];
  f.u[1] = p[1];
  return f.v;
}

/* ---- GEMM1: partial[split] = A_chunk @ W1_chunk (bf16 WMMA, double-buffered) */
__global__ __launch_bounds__(256) void gemm1_kernel(const float* __restrict__ A,
                                                    const float* __restrict__ W1,
                                                    float* __restrict__ Cpart) {
  __shared__ __bf16 sA[2][128 * 32];   /* [row][k]  2 x 8KB */
  __shared__ __bf16 sB[2][128 * 32];   /* [n][k]    2 x 8KB (transposed) */
  const int nBase = blockIdx.x * 128;
  const int mBase = blockIdx.y * 128;
  const int split = blockIdx.z;
  const int kStart = split * KCHUNK;
  const int tid  = threadIdx.x;
  const int lane = tid & 31;
  const int wave = tid >> 5;
  const int wm = wave & 3;                  /* 4 strips of 32 rows */
  const int wn = wave >> 2;                 /* 2 strips of 64 cols */
  const int row = lane & 15;
  const int hi  = lane >> 4;

  /* per-thread staging coordinates */
  const int ar  = tid >> 1;                 /* A row 0..127   */
  const int akh = (tid & 1) * 16;           /* A k sub-offset */
  const int bn  = tid & 127;                /* B col 0..127   */
  const int bkh = (tid >> 7) * 16;          /* B k sub-offset */
  const float* aSrcBase = A + (size_t)(mBase + ar) * D_IN + kStart + akh;
  const float* bSrcBase = W1 + (size_t)kStart * H1 + nBase + bn;

  v8f acc[2][4] = {};
  float4 aR[4];
  float  bR[16];

  /* prologue: stage tile 0 into buffer 0 */
  {
    #pragma unroll
    for (int q = 0; q < 4; ++q) aR[q] = ((const float4*)aSrcBase)[q];
    const float* bs = bSrcBase + (size_t)bkh * H1;
    #pragma unroll
    for (int q = 0; q < 16; ++q) bR[q] = bs[(size_t)q * H1];
    __bf16* da = sA[0] + ar * 32 + akh;
    #pragma unroll
    for (int q = 0; q < 4; ++q) {
      da[q * 4 + 0] = f2bf(aR[q].x); da[q * 4 + 1] = f2bf(aR[q].y);
      da[q * 4 + 2] = f2bf(aR[q].z); da[q * 4 + 3] = f2bf(aR[q].w);
    }
    __bf16* db = sB[0] + bn * 32 + bkh;
    #pragma unroll
    for (int q = 0; q < 16; ++q) db[q] = f2bf(bR[q]);
  }
  __syncthreads();

  const int NIT = KCHUNK / 32;              /* 98 iterations */
  for (int it = 0; it < NIT; ++it) {
    const int p = it & 1;
    const bool more = (it + 1) < NIT;

    /* issue global loads for tile it+1 before the matrix ops */
    if (more) {
      const int kb = (it + 1) * 32;
      const float* as = aSrcBase + kb;
      #pragma unroll
      for (int q = 0; q < 4; ++q) aR[q] = ((const float4*)as)[q];
      const float* bs = bSrcBase + (size_t)(kb + bkh) * H1;
      #pragma unroll
      for (int q = 0; q < 16; ++q) bR[q] = bs[(size_t)q * H1];
      if (it + 8 < NIT) {                   /* L2 stream prefetch ~8 tiles ahead */
        __builtin_prefetch(aSrcBase + (size_t)(it + 8) * 32, 0, 3);
        __builtin_prefetch(bSrcBase + (size_t)((it + 8) * 32 + bkh) * H1, 0, 3);
      }
    }

    v16bf aF[2], bF[4];
    #pragma unroll
    for (int i = 0; i < 2; ++i) aF[i] = load_frag_A(sA[p], wm * 32 + i * 16 + row, hi);
    #pragma unroll
    for (int j = 0; j < 4; ++j) bF[j] = load_frag_B(sB[p], wn * 64 + j * 16 + row, hi * 16);
    #pragma unroll
    for (int i = 0; i < 2; ++i)
      #pragma unroll
      for (int j = 0; j < 4; ++j)
        acc[i][j] = __builtin_amdgcn_wmma_f32_16x16x32_bf16(
            false, aF[i], false, bF[j], (short)0, acc[i][j], false, false);

    /* convert + store next tile into the other buffer */
    if (more) {
      __bf16* da = sA[1 - p] + ar * 32 + akh;
      #pragma unroll
      for (int q = 0; q < 4; ++q) {
        da[q * 4 + 0] = f2bf(aR[q].x); da[q * 4 + 1] = f2bf(aR[q].y);
        da[q * 4 + 2] = f2bf(aR[q].z); da[q * 4 + 3] = f2bf(aR[q].w);
      }
      __bf16* db = sB[1 - p] + bn * 32 + bkh;
      #pragma unroll
      for (int q = 0; q < 16; ++q) db[q] = f2bf(bR[q]);
    }
    __syncthreads();                        /* single barrier per K-step */
  }

  float* out = Cpart + (size_t)split * BATCH * H1;
  #pragma unroll
  for (int i = 0; i < 2; ++i)
    #pragma unroll
    for (int j = 0; j < 4; ++j) {
      const int m0 = mBase + wm * 32 + i * 16 + hi * 8;
      const int n0 = nBase + wn * 64 + j * 16 + row;
      #pragma unroll
      for (int v = 0; v < 8; ++v)
        out[(size_t)(m0 + v) * H1 + n0] = acc[i][j][v];
    }
}

/* ---------------- reduce splits + bias + relu -> h1 (bf16) ------------------ */
__global__ __launch_bounds__(256) void reduce1_kernel(const float* __restrict__ Cpart,
                                                      const float* __restrict__ b1,
                                                      __bf16* __restrict__ h1bf) {
  const int i = (blockIdx.x * 256 + threadIdx.x) * 4;    /* 262144 total elems */
  float4 s = *(const float4*)(b1 + (i & (H1 - 1)));
  #pragma unroll
  for (int p = 0; p < SPLITS; ++p) {
    float4 c = *(const float4*)(Cpart + (size_t)p * BATCH * H1 + i);
    s.x += c.x; s.y += c.y; s.z += c.z; s.w += c.w;
  }
  h1bf[i + 0] = f2bf(fmaxf(s.x, 0.0f));
  h1bf[i + 1] = f2bf(fmaxf(s.y, 0.0f));
  h1bf[i + 2] = f2bf(fmaxf(s.z, 0.0f));
  h1bf[i + 3] = f2bf(fmaxf(s.w, 0.0f));
}

/* ---------------- GEMM2: h2 = relu(h1 @ W2 + b2) (bf16 WMMA) ---------------- */
__global__ __launch_bounds__(256) void gemm2_kernel(const __bf16* __restrict__ Abf,
                                                    const float* __restrict__ W2,
                                                    const float* __restrict__ b2,
                                                    float* __restrict__ h2) {
  __shared__ __bf16 sA[128 * 32];
  __shared__ __bf16 sB[128 * 32];
  const int nBase = blockIdx.x * 128;   /* gridDim.x = 2 */
  const int mBase = blockIdx.y * 128;   /* gridDim.y = 4 */
  const int tid  = threadIdx.x;
  const int lane = tid & 31;
  const int wave = tid >> 5;
  const int wm = wave & 3, wn = wave >> 2;
  const int row = lane & 15, hi = lane >> 4;

  v8f acc[2][4] = {};

  for (int kb = 0; kb < H1; kb += 32) {
    { /* A already bf16 */
      const int r  = tid >> 1;
      const int kh = (tid & 1) * 16;
      const uint4* src = (const uint4*)(Abf + (size_t)(mBase + r) * H1 + kb + kh);
      uint4* dst = (uint4*)(sA + r * 32 + kh);
      dst[0] = src[0];
      dst[1] = src[1];
    }
    { /* B transposed, f32 -> bf16 */
      const int n  = tid & 127;
      const int kh = (tid >> 7) * 16;
      const float* src = W2 + (size_t)(kb + kh) * H2 + (nBase + n);
      __bf16* dst = sB + n * 32 + kh;
      #pragma unroll
      for (int q = 0; q < 16; ++q) dst[q] = f2bf(src[(size_t)q * H2]);
    }
    __syncthreads();

    v16bf aF[2], bF[4];
    #pragma unroll
    for (int i = 0; i < 2; ++i) aF[i] = load_frag_A(sA, wm * 32 + i * 16 + row, hi);
    #pragma unroll
    for (int j = 0; j < 4; ++j) bF[j] = load_frag_B(sB, wn * 64 + j * 16 + row, hi * 16);
    #pragma unroll
    for (int i = 0; i < 2; ++i)
      #pragma unroll
      for (int j = 0; j < 4; ++j)
        acc[i][j] = __builtin_amdgcn_wmma_f32_16x16x32_bf16(
            false, aF[i], false, bF[j], (short)0, acc[i][j], false, false);
    __syncthreads();
  }

  #pragma unroll
  for (int i = 0; i < 2; ++i)
    #pragma unroll
    for (int j = 0; j < 4; ++j) {
      const int m0 = mBase + wm * 32 + i * 16 + hi * 8;
      const int n0 = nBase + wn * 64 + j * 16 + row;
      const float bias = b2[n0];
      #pragma unroll
      for (int v = 0; v < 8; ++v)
        h2[(size_t)(m0 + v) * H2 + n0] = fmaxf(acc[i][j][v] + bias, 0.0f);
    }
}

/* ---------------- GEMM3 + sigmoid (tiny, scalar f32) ------------------------ */
__global__ __launch_bounds__(256) void gemm3_kernel(const float* __restrict__ h2,
                                                    const float* __restrict__ W3,
                                                    const float* __restrict__ b3,
                                                    float* __restrict__ pred) {
  const int t = blockIdx.x * 256 + threadIdx.x;          /* 8192 total */
  const int b = t >> 4, o = t & 15;
  float s = b3[o];
  const float* hr = h2 + (size_t)b * H2;
  #pragma unroll 4
  for (int k = 0; k < H2; ++k) s += hr[k] * W3[k * NOUT + o];
  pred[t] = 1.0f / (1.0f + __expf(-s));
}

/* ---------------- patch gather (coords -> 6x6 window) ----------------------- */
__device__ __forceinline__ int clip_c(float v) {
  int c = (int)(v * 224.0f);     /* trunc toward zero, matches astype(int32) */
  if (c >= 219) c = 200;
  if (c <= 5)   c = 8;
  return c;
}

__global__ __launch_bounds__(256) void gather_kernel(const float* __restrict__ pred_map,
                                                     const float* __restrict__ lm,
                                                     float* __restrict__ outp) {
  const int idx = blockIdx.x * 256 + threadIdx.x;        /* 147456 total */
  const int b   = idx / 288;
  const int rem = idx - b * 288;
  const int l   = rem / 36;
  const int ij  = rem - l * 36;
  const int i   = ij / 6, j = ij - i * 6;
  const int x = clip_c(lm[b * 16 + l * 2 + 0]);
  const int y = clip_c(lm[b * 16 + l * 2 + 1]);
  outp[idx] = pred_map[(size_t)b * D_IN + (size_t)(x + i - 3) * 224 + (y + j - 3)];
}

extern "C" void kernel_launch(void* const* d_in, const int* in_sizes, int n_in,
                              void* d_out, int out_size, void* d_ws, size_t ws_size,
                              hipStream_t stream) {
  const float* pred_map = (const float*)d_in[0];
  const float* gt = (const float*)d_in[1];
  const float* W1 = (const float*)d_in[2];
  const float* b1 = (const float*)d_in[3];
  const float* W2 = (const float*)d_in[4];
  const float* b2 = (const float*)d_in[5];
  const float* W3 = (const float*)d_in[6];
  const float* b3 = (const float*)d_in[7];
  float* out = (float*)d_out;

  char* ws = (char*)d_ws;
  float* part = (float*)(ws + WS_PART_OFF);
  __bf16* h1bf = (__bf16*)(ws + WS_H1_OFF);
  float* h2 = (float*)(ws + WS_H2_OFF);

  /* h1 partials: 4x4 output tiles x 16 K-splits = 256 workgroups */
  gemm1_kernel<<<dim3(4, 4, SPLITS), 256, 0, stream>>>(pred_map, W1, part);
  reduce1_kernel<<<(BATCH * H1 / 4) / 256, 256, 0, stream>>>(part, b1, h1bf);
  gemm2_kernel<<<dim3(2, 4), 256, 0, stream>>>(h1bf, W2, b2, h2);
  gemm3_kernel<<<(BATCH * NOUT) / 256, 256, 0, stream>>>(h2, W3, b3, out);

  /* out layout: pred[8192] | GT_patch[147456] | Pred_patch[147456] */
  gather_kernel<<<(BATCH * 288) / 256, 256, 0, stream>>>(pred_map, gt, out + 8192);
  gather_kernel<<<(BATCH * 288) / 256, 256, 0, stream>>>(pred_map, out,
                                                         out + 8192 + BATCH * 288);
}
